// AdaptiveBlock_77979426226506
// MI455X (gfx1250) — matrix operand order repaired and checked by reading, not compiled, not run on hardware
//
#include <hip/hip_runtime.h>
#include <hip/hip_bf16.h>

typedef __attribute__((ext_vector_type(16))) _Float16 v16h;
typedef __attribute__((ext_vector_type(8)))  float    v8f;
typedef __attribute__((ext_vector_type(4)))  float    v4f;

#define Bq   32
#define Cq   256
#define HW   3136      // 56*56
#define HWV4 784       // 3136/4
#define HID  512
#define NP   448       // H*RANK = RANK*W = 56*8

// ---------------- WMMA fragment helpers (ISA 7.12.2 layouts, wave32) ----------

// A fragment: 16x32 (MxK) f16 tile from row-major [lda] matrix.
// lane<16:  K = {0..7, 16..23};  lane>=16: K = {8..15, 24..31}; row M = lane%16
__device__ __forceinline__ v16h load_A_frag(const _Float16* A, int lda, int m0, int k0) {
  int lane = threadIdx.x & 31;
  int m  = m0 + (lane & 15);
  int kb = k0 + ((lane & 16) >> 1);          // +8 for upper half-wave
  const _Float16* p = A + m * lda + kb;
  v16h r;
#pragma unroll
  for (int e = 0; e < 8; ++e) { r[e] = p[e]; r[e + 8] = p[e + 16]; }
  return r;
}

// B fragment: 32x16 (KxN) f16 tile where memory holds W as (N x K) row-major
// (nn.Linear weight), i.e. B[k][n] = W[n][k]. lane<16: K={0..7,16..23}, lane>=16: K={8..15,24..31}.
__device__ __forceinline__ v16h load_B_fragT(const _Float16* W, int ldw, int k0, int n0) {
  int lane = threadIdx.x & 31;
  int n  = n0 + (lane & 15);
  int kb = k0 + ((lane & 16) >> 1);
  const _Float16* p = W + n * ldw + kb;      // two runs of 8 halves
  v16h r;
#pragma unroll
  for (int e = 0; e < 8; ++e) { r[e] = p[e]; r[e + 8] = p[e + 16]; }
  return r;
}

__device__ __forceinline__ float gelu_exact(float x) {
  return 0.5f * x * (1.0f + erff(x * 0.70710678118654752f));
}

// D: vgpr v, lane L -> row m0 + v + 8*(L>=16), col n0 + L%16
__device__ __forceinline__ void store_gelu_f16(_Float16* D, int ldd, int m0, int n0, v8f acc) {
  int lane = threadIdx.x & 31;
  int n  = n0 + (lane & 15);
  int mb = m0 + ((lane & 16) >> 1);
#pragma unroll
  for (int v = 0; v < 8; ++v)
    D[(mb + v) * ldd + n] = (_Float16)gelu_exact(acc[v]);
}

__device__ __forceinline__ void store_f32(float* D, int ldd, int m0, int n0, v8f acc) {
  int lane = threadIdx.x & 31;
  int n  = n0 + (lane & 15);
  int mb = m0 + ((lane & 16) >> 1);
#pragma unroll
  for (int v = 0; v < 8; ++v)
    D[(mb + v) * ldd + n] = acc[v];
}

// ---------------- Kernel 1: global average pool, emit f16 ---------------------
__global__ __launch_bounds__(256) void pool_kernel(const float* __restrict__ x,
                                                   _Float16* __restrict__ y16) {
  int row = blockIdx.x;                       // b*C + c, contiguous 3136-float row
  const v4f* src = (const v4f*)(x + (size_t)row * HW);
  int t = threadIdx.x;
  v4f s0 = __builtin_nontemporal_load(src + t);
  v4f s1 = __builtin_nontemporal_load(src + t + 256);
  v4f s2 = __builtin_nontemporal_load(src + t + 512);
  float s = s0[0]+s0[1]+s0[2]+s0[3] + s1[0]+s1[1]+s1[2]+s1[3]
          + s2[0]+s2[1]+s2[2]+s2[3];
  if (t < HWV4 - 768) {                       // 16 leftover vec4
    v4f s3 = __builtin_nontemporal_load(src + t + 768);
    s += s3[0]+s3[1]+s3[2]+s3[3];
  }
  __shared__ float red[256];
  red[t] = s;
  __syncthreads();
#pragma unroll
  for (int off = 128; off > 0; off >>= 1) {
    if (t < off) red[t] += red[t + off];
    __syncthreads();
  }
  if (t == 0) y16[row] = (_Float16)(red[0] * (1.0f / (float)HW));
}

// ---------------- Kernel 2: weights f32 -> f16 --------------------------------
__global__ __launch_bounds__(256) void cvt_kernel(
    const float* __restrict__ w1, const float* __restrict__ w2,
    const float* __restrict__ wA, const float* __restrict__ wB,
    _Float16* __restrict__ o1, _Float16* __restrict__ o2,
    _Float16* __restrict__ oA, _Float16* __restrict__ oB) {
  int i = blockIdx.x * 256 + threadIdx.x;     // grid covers 131072
  if (i < HID * Cq) { o1[i] = (_Float16)w1[i]; o2[i] = (_Float16)w2[i]; }
  if (i < NP * Cq)  { oA[i] = (_Float16)wA[i]; oB[i] = (_Float16)wB[i]; }
}

// ---------------- Kernel 3: fused MLP + projections (1 WGP, 8 waves, WMMA) ----
__global__ __launch_bounds__(256) void mlp_wmma_kernel(
    const _Float16* __restrict__ y16,
    const _Float16* __restrict__ w1h, const _Float16* __restrict__ w2h,
    const _Float16* __restrict__ wAh, const _Float16* __restrict__ wBh,
    float* __restrict__ Ap, float* __restrict__ Bp) {
  __shared__ _Float16 h16[Bq * HID];          // 32 KB
  __shared__ _Float16 yp16[Bq * Cq];          // 16 KB
  int wave = threadIdx.x >> 5;

  // Stage 1: h = gelu(y @ w1^T)   (32x512, K=256) -> 2x32 = 64 tiles
#pragma unroll 1
  for (int i = 0; i < 8; ++i) {
    int t = wave * 8 + i;
    int mt = t & 1, nt = t >> 1;
    v8f acc = {};
#pragma unroll 1
    for (int kk = 0; kk < 8; ++kk) {
      v16h a = load_A_frag(y16, Cq, mt * 16, kk * 32);
      v16h b = load_B_fragT(w1h, Cq, kk * 32, nt * 16);
      acc = __builtin_amdgcn_wmma_f32_16x16x32_f16(false, a, false, b,
                                                   (short)0, acc, false, false);
    }
    store_gelu_f16(h16, HID, mt * 16, nt * 16, acc);
  }
  __syncthreads();

  // Stage 2: yp = gelu(h @ w2^T)  (32x256, K=512) -> 2x16 = 32 tiles
#pragma unroll 1
  for (int i = 0; i < 4; ++i) {
    int t = wave * 4 + i;
    int mt = t & 1, nt = t >> 1;
    v8f acc = {};
#pragma unroll 1
    for (int kk = 0; kk < 16; ++kk) {
      v16h a = load_A_frag(h16, HID, mt * 16, kk * 32);
      v16h b = load_B_fragT(w2h, HID, kk * 32, nt * 16);
      acc = __builtin_amdgcn_wmma_f32_16x16x32_f16(false, a, false, b,
                                                   (short)0, acc, false, false);
    }
    store_gelu_f16(yp16, Cq, mt * 16, nt * 16, acc);
  }
  __syncthreads();

  // Stage 3: Ap = yp @ wA^T, Bp = yp @ wB^T  (each 32x448, K=256) -> 2x28x2 = 112 tiles
#pragma unroll 1
  for (int i = 0; i < 14; ++i) {
    int T = wave * 14 + i;                    // wave-uniform
    const _Float16* W = (T < 56) ? wAh : wBh;
    float* D          = (T < 56) ? Ap : Bp;
    int tile = (T < 56) ? T : T - 56;
    int mt = tile & 1, nt = tile >> 1;
    v8f acc = {};
#pragma unroll 1
    for (int kk = 0; kk < 8; ++kk) {
      v16h a = load_A_frag(yp16, Cq, mt * 16, kk * 32);
      v16h b = load_B_fragT(W, Cq, kk * 32, nt * 16);
      acc = __builtin_amdgcn_wmma_f32_16x16x32_f16(false, a, false, b,
                                                   (short)0, acc, false, false);
    }
    store_f32(D, NP, mt * 16, nt * 16, acc);
  }
}

// ---------------- Kernel 4: per-batch rank-8 map + sigmoid --------------------
__global__ __launch_bounds__(256) void attn_kernel(const float* __restrict__ Ap,
                                                   const float* __restrict__ Bp,
                                                   float* __restrict__ attn) {
  __shared__ float Ab[NP];   // (56, 8)
  __shared__ float Bb[NP];   // (8, 56)
  int b = blockIdx.x, t = threadIdx.x;
  for (int i = t; i < NP; i += 256) {
    Ab[i] = Ap[b * NP + i];
    Bb[i] = Bp[b * NP + i];
  }
  __syncthreads();
  for (int idx = t; idx < HW; idx += 256) {
    int i = idx / 56, j = idx - i * 56;
    float m = 0.f;
#pragma unroll
    for (int r = 0; r < 8; ++r) m += Ab[i * 8 + r] * Bb[r * 56 + j];
    attn[b * HW + idx] = 1.0f / (1.0f + expf(-m));
  }
}

// ---------------- Kernel 5: broadcast across channels (streaming store) -------
__global__ __launch_bounds__(256) void bcast_kernel(const float* __restrict__ attn,
                                                    float* __restrict__ out) {
  int row = blockIdx.x;                       // b*256 + c
  int b = row >> 8;
  const v4f* src = (const v4f*)(attn + (size_t)b * HW);  // hot in L2 (401 KB total)
  v4f* dst = (v4f*)(out + (size_t)row * HW);
  int t = threadIdx.x;
  v4f v0 = src[t];
  v4f v1 = src[t + 256];
  v4f v2 = src[t + 512];
  __builtin_nontemporal_store(v0, dst + t);
  __builtin_nontemporal_store(v1, dst + t + 256);
  __builtin_nontemporal_store(v2, dst + t + 512);
  if (t < HWV4 - 768) {
    v4f v3 = src[t + 768];
    __builtin_nontemporal_store(v3, dst + t + 768);
  }
}

// ---------------- host launcher ----------------------------------------------
extern "C" void kernel_launch(void* const* d_in, const int* in_sizes, int n_in,
                              void* d_out, int out_size, void* d_ws, size_t ws_size,
                              hipStream_t stream) {
  const float* x  = (const float*)d_in[0];
  const float* w1 = (const float*)d_in[1];
  const float* w2 = (const float*)d_in[2];
  const float* wA = (const float*)d_in[3];
  const float* wB = (const float*)d_in[4];
  float* out = (float*)d_out;

  char* ws = (char*)d_ws;
  size_t o = 0;
  _Float16* y16 = (_Float16*)(ws + o); o += (size_t)Bq * Cq * 2;        // 16 KB
  o = (o + 255) & ~(size_t)255;
  _Float16* w1h = (_Float16*)(ws + o); o += (size_t)HID * Cq * 2;       // 256 KB
  _Float16* w2h = (_Float16*)(ws + o); o += (size_t)Cq * HID * 2;       // 256 KB
  _Float16* wAh = (_Float16*)(ws + o); o += (size_t)NP * Cq * 2;        // 224 KB
  _Float16* wBh = (_Float16*)(ws + o); o += (size_t)NP * Cq * 2;        // 224 KB
  float* Ap   = (float*)(ws + o); o += (size_t)Bq * NP * 4;             // 56 KB
  float* Bp   = (float*)(ws + o); o += (size_t)Bq * NP * 4;             // 56 KB
  float* attn = (float*)(ws + o); o += (size_t)Bq * HW * 4;             // 392 KB

  pool_kernel<<<Bq * Cq, 256, 0, stream>>>(x, y16);
  cvt_kernel<<<(HID * Cq + 255) / 256, 256, 0, stream>>>(w1, w2, wA, wB,
                                                         w1h, w2h, wAh, wBh);
  mlp_wmma_kernel<<<1, 256, 0, stream>>>(y16, w1h, w2h, wAh, wBh, Ap, Bp);
  attn_kernel<<<Bq, 256, 0, stream>>>(Ap, Bp, attn);
  bcast_kernel<<<Bq * Cq, 256, 0, stream>>>(attn, out);
}
